// ParallelHSRnnV2_59072980189823
// MI455X (gfx1250) — matrix operand order repaired and checked by reading, not compiled, not run on hardware
//
#include <hip/hip_runtime.h>
#include <hip/hip_bf16.h>
#include <math.h>

// ---------------------------------------------------------------------------
// Types for CDNA5 WMMA (gfx1250, wave32)
// ---------------------------------------------------------------------------
typedef __bf16 bf16_t;
typedef __attribute__((ext_vector_type(16))) __bf16 v16bf;
typedef __attribute__((ext_vector_type(8)))  __bf16 v8bf;
typedef __attribute__((ext_vector_type(4)))  __bf16 v4bf;
typedef __attribute__((ext_vector_type(8)))  float  v8f;
typedef __attribute__((ext_vector_type(4)))  float  v4f;

#define BM 128
#define BN 128
#define BK 32
#define BKP 40   // padded LDS stride (halves): 80B rows -> conflict-free b128 reads

static __device__ __forceinline__ v16bf cat8(v8bf lo, v8bf hi) {
  return __builtin_shufflevector(lo, hi, 0,1,2,3,4,5,6,7,8,9,10,11,12,13,14,15);
}

// A-matrix 16x32 bf16 fragment: lanes 0-15 -> M=lane; half-wave selects K-phase.
static __device__ __forceinline__ v16bf load_a_frag(const bf16_t* row, int lsel) {
  v8bf lo = *(const v8bf*)(row + lsel * 8);
  v8bf hi = *(const v8bf*)(row + 16 + lsel * 8);
  return cat8(lo, hi);
}

// B-matrix 32x16 bf16 fragment (N-major in LDS): lane n = l&15; half-wave selects K-phase.
static __device__ __forceinline__ v16bf load_b_frag(const bf16_t* row, int lsel) {
  v8bf lo = *(const v8bf*)(row + lsel * 8);
  v8bf hi = *(const v8bf*)(row + 16 + lsel * 8);
  return cat8(lo, hi);
}

// Generic->LDS offset: hardware maps flat LDS addresses via addr[31:0].
static __device__ __forceinline__ uint32_t lds_addr_u32(const void* p) {
  return (uint32_t)(size_t)p;
}

// CDNA5 async copy: global -> LDS, 16B per lane, tracked by ASYNCcnt.
static __device__ __forceinline__ void async_copy_b128(uint32_t lds, uint32_t goff,
                                                       const void* gbase) {
  asm volatile("global_load_async_to_lds_b128 %0, %1, %2 offset:0"
               :: "v"(lds), "v"(goff), "s"(gbase) : "memory");
}

// ---------------------------------------------------------------------------
// fp32 -> bf16 bulk conversion (8 elements / thread, b128 stores)
// ---------------------------------------------------------------------------
__global__ __launch_bounds__(256) void cvt_f32_bf16(
    const float* __restrict__ in, bf16_t* __restrict__ out, long n8)
{
  long i = (long)blockIdx.x * blockDim.x + threadIdx.x;
  if (i >= n8) return;
  const v4f* p = (const v4f*)in + i * 2;
  v4f a = p[0], b = p[1];
  v8bf o;
#pragma unroll
  for (int j = 0; j < 4; ++j) { o[j] = (bf16_t)a[j]; o[4 + j] = (bf16_t)b[j]; }
  ((v8bf*)out)[i] = o;
}

// ---------------------------------------------------------------------------
// GEMM:  C[m][n] = epilogue( sum_k A[m][k] * W[n][k] + bias[n] )
//   A: [M,K] bf16 row-major;  W: [N,K] bf16 row-major (acts as B^T);  C fp32.
// 256 threads = 8 waves; 128x128x32 tile; double-buffered LDS filled by
// GLOBAL_LOAD_ASYNC_TO_LDS_B128; pipeline unrolled x2 so buffer indices are
// compile-time constants (constant ds_load offsets, no select logic).
// ---------------------------------------------------------------------------
template <bool TANH_OUT>
__global__ __launch_bounds__(256) void gemm_bias_wmma(
    const bf16_t* __restrict__ Ab, const bf16_t* __restrict__ Wb,
    const float* __restrict__ bias, float* __restrict__ Cp,
    int M, int N, int K)
{
  __shared__ alignas(16) bf16_t As[2][BM * BKP];
  __shared__ alignas(16) bf16_t Bs[2][BN * BKP];

  const int tid  = threadIdx.x;
  const int lane = tid & 31;
  const int wave = tid >> 5;
  const int lrow = lane & 15;
  const int lsel = lane >> 4;

  const int m0 = blockIdx.x * BM;
  const int n0 = blockIdx.y * BN;
  const int wm0 = (wave & 1) * 64;   // 2 waves along M
  const int wn0 = (wave >> 1) * 32;  // 4 waves along N

  // Per-thread staging coordinates: 2 b128 chunks for A, 2 for B, per tile.
  uint32_t goffA[2], goffB[2], ldsA[2][2], ldsB[2][2];
#pragma unroll
  for (int i = 0; i < 2; ++i) {
    const int idx = i * 256 + tid;          // 512 chunks of 8 halves = 128x32
    const int row = idx >> 2, c8 = idx & 3;
    goffA[i] = (uint32_t)(((size_t)(m0 + row) * K + c8 * 8) * sizeof(bf16_t));
    goffB[i] = (uint32_t)(((size_t)(n0 + row) * K + c8 * 8) * sizeof(bf16_t));
#pragma unroll
    for (int b = 0; b < 2; ++b) {
      ldsA[b][i] = lds_addr_u32(&As[b][row * BKP + c8 * 8]);
      ldsB[b][i] = lds_addr_u32(&Bs[b][row * BKP + c8 * 8]);
    }
  }

  v8f acc[4][2];
  v8f vzero;
#pragma unroll
  for (int i = 0; i < 8; ++i) vzero[i] = 0.0f;
#pragma unroll
  for (int mi = 0; mi < 4; ++mi)
#pragma unroll
    for (int ni = 0; ni < 2; ++ni) acc[mi][ni] = vzero;

  // issue one tile's async copies into buffer `buf` (4 ops per wave)
  auto issue_tile = [&](int kt, int buf) {
    const uint32_t kb = (uint32_t)(kt * BK * sizeof(bf16_t));
#pragma unroll
    for (int i = 0; i < 2; ++i) async_copy_b128(ldsA[buf][i], goffA[i] + kb, Ab);
#pragma unroll
    for (int i = 0; i < 2; ++i) async_copy_b128(ldsB[buf][i], goffB[i] + kb, Wb);
  };

  // consume buffer `buf` (compile-time constant) with 8 WMMAs
  auto compute_tile = [&](int buf) {
    v16bf afr[4], bfr[2];
#pragma unroll
    for (int mi = 0; mi < 4; ++mi)
      afr[mi] = load_a_frag(&As[buf][(wm0 + mi * 16 + lrow) * BKP], lsel);
#pragma unroll
    for (int ni = 0; ni < 2; ++ni)
      bfr[ni] = load_b_frag(&Bs[buf][(wn0 + ni * 16 + lrow) * BKP], lsel);
#pragma unroll
    for (int mi = 0; mi < 4; ++mi)
#pragma unroll
      for (int ni = 0; ni < 2; ++ni)
        acc[mi][ni] = __builtin_amdgcn_wmma_f32_16x16x32_bf16(
            false, afr[mi], false, bfr[ni], (short)0, acc[mi][ni], false, false);
  };

  const int ktiles = K / BK;              // even (K % 64 == 0 here)
  issue_tile(0, 0);                       // prologue: fill buffer 0

#pragma unroll 1
  for (int kt = 0; kt < ktiles; kt += 2) {
    // ---- buffer 0 ----
    issue_tile(kt + 1, 1);                               // overlap next copy
    asm volatile("s_wait_asynccnt 0x4" ::: "memory");    // tile kt landed (this wave)
    __syncthreads();
    compute_tile(0);
    __syncthreads();                                     // done reading buf 0

    // ---- buffer 1 ----
    if (kt + 2 < ktiles) {
      issue_tile(kt + 2, 0);
      asm volatile("s_wait_asynccnt 0x4" ::: "memory");
    } else {
      asm volatile("s_wait_asynccnt 0x0" ::: "memory");
    }
    __syncthreads();
    compute_tile(1);
    __syncthreads();                                     // done reading buf 1
  }

  // Epilogue: D element (VGPR r, lane l): m = r + 8*(l>>4), n = l&15
#pragma unroll
  for (int ni = 0; ni < 2; ++ni) {
    const int n = n0 + wn0 + ni * 16 + lrow;
    const float bv = bias[n];
#pragma unroll
    for (int mi = 0; mi < 4; ++mi) {
      const int mbase = m0 + wm0 + mi * 16 + 8 * lsel;
#pragma unroll
      for (int r = 0; r < 8; ++r) {
        float v = acc[mi][ni][r] + bv;
        if (TANH_OUT) v = tanhf(v);
        Cp[(size_t)(mbase + r) * N + n] = v;
      }
    }
  }
}

// ---------------------------------------------------------------------------
// HSRU leaky-integrate / threshold-reset scan. One thread per (b,h), coalesced
// in h. Loads for t+PF..t+2PF-1 are independent of the recurrence, so they are
// prefetched into registers to hide memory latency at low wave occupancy.
// ---------------------------------------------------------------------------
#define PF 16
__global__ __launch_bounds__(256) void hsru_scan(
    const float* __restrict__ I, const float* __restrict__ leak_tau,
    const float* __restrict__ thr, bf16_t* __restrict__ comb,
    int Bdim, int T, int H)
{
  const int gid = blockIdx.x * blockDim.x + threadIdx.x;
  if (gid >= Bdim * H) return;
  const int b = gid / H;
  const int h = gid - b * H;

  const float lt = leak_tau[h];
  const float sp = (lt > 20.0f) ? lt : log1pf(expf(lt));   // softplus
  const float alpha = expf(-sp);
  const float th = thr[h];

  const float* Ip = I + (size_t)b * T * H + h;
  bf16_t* cb = comb + (size_t)b * T * (2 * H) + h;
  const size_t cstride = 2 * (size_t)H;

  float nxt[PF];
#pragma unroll
  for (int j = 0; j < PF; ++j) nxt[j] = Ip[(size_t)j * H];

  float v = 0.0f;
  for (int t = 0; t < T; t += PF) {
    float cur[PF];
#pragma unroll
    for (int j = 0; j < PF; ++j) cur[j] = nxt[j];
    if (t + PF < T) {
#pragma unroll
      for (int j = 0; j < PF; ++j) nxt[j] = Ip[(size_t)(t + PF + j) * H];
    }
#pragma unroll
    for (int j = 0; j < PF; ++j) {
      v = fmaf(alpha, v, cur[j]);
      const bool fire = (v > th);
      v = fire ? 0.0f : v;                        // post-reset membrane
      cb[(size_t)(t + j) * cstride]     = (bf16_t)v;
      cb[(size_t)(t + j) * cstride + H] = (bf16_t)(fire ? 1.0f : 0.0f);
    }
  }
}

// ---------------------------------------------------------------------------
// Host launcher
// ---------------------------------------------------------------------------
extern "C" void kernel_launch(void* const* d_in, const int* in_sizes, int n_in,
                              void* d_out, int out_size, void* d_ws, size_t ws_size,
                              hipStream_t stream)
{
  const float* x      = (const float*)d_in[0];  // [B,T,IN]
  const float* W_in   = (const float*)d_in[1];  // [H,IN]
  const float* b_in   = (const float*)d_in[2];  // [H]
  const float* leak   = (const float*)d_in[3];  // [H]
  const float* thresh = (const float*)d_in[4];  // [H]
  const float* W_out  = (const float*)d_in[5];  // [H,2H]
  const float* b_out  = (const float*)d_in[6];  // [H]

  const int H  = in_sizes[2];
  const int IN = in_sizes[1] / H;
  const int BT = in_sizes[0] / IN;      // B*T
  const int K2 = in_sizes[5] / H;       // 2H
  const int B  = 16;
  const int T  = BT / B;

  // Workspace layout (bytes):
  char* ws = (char*)d_ws;
  float*  Ibuf = (float*)ws;                                  // [BT,H]  fp32
  ws += (size_t)BT * H * sizeof(float);
  bf16_t* comb = (bf16_t*)ws;                                 // [BT,2H] bf16
  ws += (size_t)BT * K2 * sizeof(bf16_t);
  bf16_t* xb   = (bf16_t*)ws;                                 // [BT,IN] bf16
  ws += (size_t)BT * IN * sizeof(bf16_t);
  bf16_t* wib  = (bf16_t*)ws;                                 // [H,IN]  bf16
  ws += (size_t)H * IN * sizeof(bf16_t);
  bf16_t* wob  = (bf16_t*)ws;                                 // [H,2H]  bf16

  float* out = (float*)d_out;                                 // [BT,H]  fp32

  // 0) one-shot fp32 -> bf16 conversion of GEMM operands
  {
    long n8;
    n8 = (long)BT * IN / 8;
    cvt_f32_bf16<<<(unsigned)((n8 + 255) / 256), 256, 0, stream>>>(x, xb, n8);
    n8 = (long)H * IN / 8;
    cvt_f32_bf16<<<(unsigned)((n8 + 255) / 256), 256, 0, stream>>>(W_in, wib, n8);
    n8 = (long)H * K2 / 8;
    cvt_f32_bf16<<<(unsigned)((n8 + 255) / 256), 256, 0, stream>>>(W_out, wob, n8);
  }

  // 1) I = x @ W_in^T + b_in
  dim3 g1(BT / BM, H / BN);
  gemm_bias_wmma<false><<<g1, 256, 0, stream>>>(xb, wib, b_in, Ibuf, BT, H, IN);

  // 2) sequential scan over T, emit combined [BT, 2H] bf16
  const int nscan = B * H;
  hsru_scan<<<(nscan + 255) / 256, 256, 0, stream>>>(
      Ibuf, leak, thresh, comb, B, T, H);

  // 3) out = tanh(combined @ W_out^T + b_out)
  dim3 g2(BT / BM, H / BN);
  gemm_bias_wmma<true><<<g2, 256, 0, stream>>>(comb, wob, b_out, out, BT, H, K2);
}
